// QRNN_9405978378440
// MI455X (gfx1250) — matrix operand order repaired and checked by reading, not compiled, not run on hardware
//
#include <hip/hip_runtime.h>
#include <hip/hip_bf16.h>
#include <math.h>

// ---------------------------------------------------------------------------
// QRNN fo-pooling for MI455X (gfx1250, wave32).  B=16,T=1024,D=512,H=512,K=2
//  1) cvt_x        : x fp32 -> bf16 (conv k=2 rows are 1024 contiguous elems)
//  2) pack_w       : W[H,D,2] -> Wpack[1536 x 1024] bf16 row-major [n,k]
//  3) gemm_act     : M=16368,K=1024,N=1536 via v_wmma_f32_16x16x32_bf16.
//                    32Mx64N per wave, register double-buffered K pipeline
//                    (next-k loads issued before current-k WMMAs -> partial
//                    s_wait_loadcnt instead of full stalls). Fused bias+act.
//  4) chunked scan : c'=f*c+(1-f)z is affine => compose per 64-step chunk:
//                    pass1 (P,Q) per chunk (4096 waves), pass2 16-step serial
//                    combine, pass3 replay+write. 16x less serial latency.
// ---------------------------------------------------------------------------

typedef __bf16 bf16_t;
typedef __bf16 v16bf __attribute__((ext_vector_type(16)));
typedef __bf16 v8bf  __attribute__((ext_vector_type(8)));
typedef __bf16 v4bf  __attribute__((ext_vector_type(4)));
typedef float  v8f   __attribute__((ext_vector_type(8)));

#define B_  16
#define T_  1024
#define D_  512
#define H_  512
#define TM  (T_ - 1)        // 1023
#define KG  (2 * D_)        // 1024 (GEMM K)
#define MG  (B_ * TM)       // 16368 (GEMM M)
#define GATE_ELEMS ((size_t)MG * H_)
#define NCH 16              // scan chunks
#define CH  64              // steps per chunk (last chunk: 63)
#define BH  (B_ * H_)       // 8192 channels

// ---------------------------- 1) x -> bf16 ---------------------------------
__global__ void cvt_x(const float4* __restrict__ x, bf16_t* __restrict__ xb) {
    int i = blockIdx.x * 256 + threadIdx.x;        // < B*T*D/4
    float4 v = x[i];
    v4bf o;
    o[0] = (bf16_t)v.x; o[1] = (bf16_t)v.y; o[2] = (bf16_t)v.z; o[3] = (bf16_t)v.w;
    *(v4bf*)(xb + 4 * (size_t)i) = o;
}

// ------------------------- 2) pack weights ---------------------------------
__global__ void pack_w(const float* __restrict__ Wz, const float* __restrict__ Wf,
                       const float* __restrict__ Wo, bf16_t* __restrict__ wp) {
    int idx = blockIdx.x * 256 + threadIdx.x;      // < 3*512*1024
    int g = idx >> 19;
    int r = idx & ((512 * 1024) - 1);
    int n = r >> 10;
    int k = r & 1023;
    const float* W = (g == 0) ? Wz : (g == 1) ? Wf : Wo;
    wp[idx] = (bf16_t)W[n * 1024 + ((k & 511) << 1) + (k >> 9)];
}

// ------------------------- 3) GEMM + activation ----------------------------
__device__ __forceinline__ v16bf ldfrag(const bf16_t* p0, const bf16_t* p1) {
    v8bf lo = *(const v8bf*)p0;
    v8bf hi = *(const v8bf*)p1;
    v16bf r;
    #pragma unroll
    for (int i = 0; i < 8; ++i) { r[i] = lo[i]; r[8 + i] = hi[i]; }
    return r;
}
__device__ __forceinline__ v16bf ldA(const bf16_t* arow, int kk, int h) {
    return ldfrag(arow + kk + 8 * h, arow + kk + 16 + 8 * h);
}
__device__ __forceinline__ v16bf ldB(const bf16_t* brow, int j, int kk, int h) {
    const bf16_t* p = brow + (size_t)(16 * j) * KG + kk + 8 * h;
    return ldfrag(p, p + 16);
}
#define WMMA_BF16(A, Bm, C) \
    __builtin_amdgcn_wmma_f32_16x16x32_bf16(false, (A), false, (Bm), (short)0, (C), false, false)

// 512 M-tiles (32 rows, last padded) x 24 N64-tiles = 12288 waves; 8/block.
__global__ __launch_bounds__(256, 1)
void gemm_act(const bf16_t* __restrict__ xb, const bf16_t* __restrict__ wp,
              const float* __restrict__ bz, const float* __restrict__ bf_,
              const float* __restrict__ bo, bf16_t* __restrict__ act) {
    const int lane = threadIdx.x & 31;
    const int warp = threadIdx.x >> 5;
    const int w    = blockIdx.x * 8 + warp;
    const int mt   = w / 24;                       // 32-row M tile
    const int nt   = w % 24;                       // 64-col N tile
    const int h    = lane >> 4;
    const int ln   = lane & 15;

    // Two A sub-tiles (rows mt*32+s*16 .. +15); clamp padded rows in-bounds.
    const bf16_t* arow[2];
    #pragma unroll
    for (int s = 0; s < 2; ++s) {
        int m = mt * 32 + s * 16 + ln;
        if (m >= MG) m = MG - 1;
        int b = m / TM;
        int t = m - b * TM;
        arow[s] = xb + (size_t)(b * T_ + t) * D_;  // 1024 contiguous bf16
    }
    const int ngl  = nt * 64;
    const int gate = ngl >> 9;
    const int nin  = ngl & 511;
    const bf16_t* __restrict__ brow = wp + (size_t)(ngl + ln) * KG;
    const float* __restrict__ bias = (gate == 0) ? bz : (gate == 1) ? bf_ : bo;

    v8f acc[2][4] = {};
    v16bf aC[2], bC[4], aN[2], bN[4];

    // Prologue: load k=0 fragments.
    #pragma unroll
    for (int s = 0; s < 2; ++s) aC[s] = ldA(arow[s], 0, h);
    #pragma unroll
    for (int j = 0; j < 4; ++j) bC[j] = ldB(brow, j, 0, h);

    // Ping-pong pipelined K loop (KG/32 = 32 steps, unrolled x2).
    for (int kk = 0; kk < KG; kk += 64) {
        #pragma unroll
        for (int s = 0; s < 2; ++s) aN[s] = ldA(arow[s], kk + 32, h);
        #pragma unroll
        for (int j = 0; j < 4; ++j) bN[j] = ldB(brow, j, kk + 32, h);
        #pragma unroll
        for (int s = 0; s < 2; ++s)
            #pragma unroll
            for (int j = 0; j < 4; ++j)
                acc[s][j] = WMMA_BF16(aC[s], bC[j], acc[s][j]);

        if (kk + 64 < KG) {
            #pragma unroll
            for (int s = 0; s < 2; ++s) aC[s] = ldA(arow[s], kk + 64, h);
            #pragma unroll
            for (int j = 0; j < 4; ++j) bC[j] = ldB(brow, j, kk + 64, h);
        }
        #pragma unroll
        for (int s = 0; s < 2; ++s)
            #pragma unroll
            for (int j = 0; j < 4; ++j)
                acc[s][j] = WMMA_BF16(aN[s], bN[j], acc[s][j]);
    }

    // Epilogue: D layout => VGPR r holds M=r+8h, N=lane%16; guard padded rows.
    bf16_t* __restrict__ gact = act + (size_t)gate * GATE_ELEMS;
    #pragma unroll
    for (int s = 0; s < 2; ++s) {
        const int m0 = mt * 32 + s * 16 + 8 * h;
        #pragma unroll
        for (int j = 0; j < 4; ++j) {
            const int n  = nin + 16 * j + ln;
            const float bv = bias[n];
            #pragma unroll
            for (int r = 0; r < 8; ++r) {
                const int mo = m0 + r;
                if (mo < MG) {
                    float v = acc[s][j][r] + bv;
                    float a = (gate == 0) ? tanhf(v) : 1.0f / (1.0f + __expf(-v));
                    gact[(size_t)mo * H_ + n] = (bf16_t)a;
                }
            }
        }
    }
}

// ------------------------- 4) chunked fo-pool scan -------------------------
// Pass 1: per (chunk, b, h) compute P=prod(f), Q = local scan from c=0.
__global__ __launch_bounds__(256)
void scan_partial(const bf16_t* __restrict__ act, float* __restrict__ P,
                  float* __restrict__ Q) {
    int idx = blockIdx.x * 256 + threadIdx.x;      // < NCH*BH = 131072
    int ch  = idx >> 13;                           // /BH
    int r   = idx & (BH - 1);
    int b   = r >> 9;
    int hh  = r & 511;
    int t0 = ch * CH, t1 = (ch == NCH - 1) ? TM : t0 + CH;
    const bf16_t* __restrict__ za = act + (size_t)(b * TM + t0) * H_ + hh;
    const bf16_t* __restrict__ fa = za + GATE_ELEMS;
    float p = 1.0f, q = 0.0f;
    for (int t = t0; t < t1; ++t) {
        float z = (float)*za;
        float f = (float)*fa;
        q = fmaf(f, q, (1.0f - f) * z);
        p *= f;
        za += H_; fa += H_;
    }
    P[idx] = p; Q[idx] = q;
}

// Pass 2: sequential combine over NCH chunks -> chunk entry states cin.
__global__ __launch_bounds__(256)
void scan_combine(const float* __restrict__ P, const float* __restrict__ Q,
                  float* __restrict__ cin) {
    int r = blockIdx.x * 256 + threadIdx.x;        // < BH
    float c = 0.0f;
    #pragma unroll
    for (int ch = 0; ch < NCH; ++ch) {
        cin[ch * BH + r] = c;
        c = fmaf(P[ch * BH + r], c, Q[ch * BH + r]);
    }
}

// Pass 3: replay each chunk with correct entry state; write h = o*c.
__global__ __launch_bounds__(256)
void scan_final(const bf16_t* __restrict__ act, const float* __restrict__ cin,
                float* __restrict__ out) {
    int idx = blockIdx.x * 256 + threadIdx.x;      // < NCH*BH
    int ch  = idx >> 13;
    int r   = idx & (BH - 1);
    int b   = r >> 9;
    int hh  = r & 511;
    int t0 = ch * CH, t1 = (ch == NCH - 1) ? TM : t0 + CH;
    const bf16_t* __restrict__ za = act + (size_t)(b * TM + t0) * H_ + hh;
    const bf16_t* __restrict__ fa = za + GATE_ELEMS;
    const bf16_t* __restrict__ oa = za + 2 * GATE_ELEMS;
    float* __restrict__ op = out + (size_t)(b * TM + t0) * H_ + hh;
    float c = cin[idx];
    for (int t = t0; t < t1; ++t) {
        float z = (float)*za;
        float f = (float)*fa;
        float o = (float)*oa;
        c = fmaf(f, c, (1.0f - f) * z);
        *op = o * c;
        za += H_; fa += H_; oa += H_; op += H_;
    }
}

// ---------------------------------------------------------------------------
extern "C" void kernel_launch(void* const* d_in, const int* in_sizes, int n_in,
                              void* d_out, int out_size, void* d_ws, size_t ws_size,
                              hipStream_t stream) {
    const float* x  = (const float*)d_in[0];
    const float* Wz = (const float*)d_in[1];
    const float* bz = (const float*)d_in[2];
    const float* Wf = (const float*)d_in[3];
    const float* bf = (const float*)d_in[4];
    const float* Wo = (const float*)d_in[5];
    const float* bo = (const float*)d_in[6];
    float* out = (float*)d_out;

    // Workspace layout (bytes):
    char* ws = (char*)d_ws;
    bf16_t* xbf = (bf16_t*)ws;                                   // 16,777,216
    bf16_t* wpk = (bf16_t*)(ws + 16777216);                      //  3,145,728
    bf16_t* act = (bf16_t*)(ws + 16777216 + 3145728);            // 50,282,496
    char*   ext = ws + 16777216 + 3145728 + 50282496;
    float*  P   = (float*)(ext);                                 // 524,288
    float*  Q   = (float*)(ext + 524288);                        // 524,288
    float*  cin = (float*)(ext + 1048576);                       // 524,288

    cvt_x<<<(B_ * T_ * D_ / 4) / 256, 256, 0, stream>>>((const float4*)x, xbf);
    pack_w<<<(3 * H_ * KG) / 256, 256, 0, stream>>>(Wz, Wf, Wo, wpk);
    gemm_act<<<12288 / 8, 256, 0, stream>>>(xbf, wpk, bz, bf, bo, act);
    scan_partial<<<(NCH * BH) / 256, 256, 0, stream>>>(act, P, Q);
    scan_combine<<<BH / 256, 256, 0, stream>>>(P, Q, cin);
    scan_final<<<(NCH * BH) / 256, 256, 0, stream>>>(act, cin, out);
}
